// NeuralODE_41137196761547
// MI455X (gfx1250) — compile-verified, matrix operand order checked by reading, not run on hardware
//
#include <hip/hip_runtime.h>

typedef _Float16 v8h  __attribute__((ext_vector_type(8)));
typedef _Float16 v16h __attribute__((ext_vector_type(16)));
typedef float    v8f  __attribute__((ext_vector_type(8)));

#define DD 64    // state dim
#define HH 128   // hidden dim
#define BLOCK 128  // 4 waves cooperating on one 16-row batch tile
#define ROWS 16

union V16 { v16h v; v8h h[2]; };

__device__ __forceinline__ float fast_tanh(float x) {
#if __has_builtin(__builtin_amdgcn_tanhf)
  return __builtin_amdgcn_tanhf(x);   // gfx1250 v_tanh_f32 (trans op)
#else
  x = fminf(15.0f, fmaxf(-15.0f, x));
  float e = __expf(2.0f * x);
  return (e - 1.0f) * __builtin_amdgcn_rcpf(e + 1.0f);
#endif
}

// Stage this wave's 16-column slice of y = z + c*k into the block-shared f16
// z-buffer (row-major [16][64]), then sync so all waves can read A-fragments.
__device__ __forceinline__ void stage(_Float16* zb, int lane, int wave,
                                      const float z[8], const float k[8],
                                      float c) {
  const int lm = lane & 15;
  const int hi = lane >> 4;
#pragma unroll
  for (int j = 0; j < 8; ++j)
    zb[(j + hi * 8) * DD + wave * 16 + lm] = (_Float16)(z[j] + c * k[j]);
  __syncthreads();
}

// One evaluation of f. Work split 4 ways: this wave computes GEMM1 N-tiles
// {2*wave, 2*wave+1} and GEMM2 N-tile {wave}; k holds this wave's 16-column
// slice of f(y) in C/D register layout.
__device__ __forceinline__ void feval(const _Float16* zb, _Float16* hb,
                                      const _Float16* w1frag,
                                      const _Float16* w2frag,
                                      const float* b1s, const float* b2s,
                                      int lane, int wave, float k[8]) {
  const int lm = lane & 15;
  const int hi = lane >> 4;

  // A fragments of y: M=16, K=64 -> 2 K-tiles of 32
  V16 a[2];
#pragma unroll
  for (int kt = 0; kt < 2; ++kt) {
    a[kt].h[0] = *(const v8h*)&zb[lm * DD + kt * 32 + hi * 8];
    a[kt].h[1] = *(const v8h*)&zb[lm * DD + kt * 32 + 16 + hi * 8];
  }

  // GEMM1 slice: two N-tiles of tanh(y@W1 + b1) -> hb
#pragma unroll
  for (int t = 0; t < 2; ++t) {
    const int nt = wave * 2 + t;
    v8f acc = {};  // SRC2 folds to inline constant 0
#pragma unroll
    for (int kt = 0; kt < 2; ++kt) {
      V16 bf;
      const _Float16* p = &w1frag[((nt * 2 + kt) * 32 + lane) * 16];
      bf.h[0] = *(const v8h*)p;
      bf.h[1] = *(const v8h*)(p + 8);
      acc = __builtin_amdgcn_wmma_f32_16x16x32_f16(
          false, a[kt].v, false, bf.v, (short)0, acc, false, false);
    }
    const float bv = b1s[nt * 16 + lm];
#pragma unroll
    for (int j = 0; j < 8; ++j)
      hb[(j + hi * 8) * HH + nt * 16 + lm] = (_Float16)fast_tanh(acc[j] + bv);
  }
  __syncthreads();  // hbuf ready; also fences zbuf reads before next stage

  // A fragments of h: M=16, K=128 -> 4 K-tiles of 32
  V16 ah[4];
#pragma unroll
  for (int kt = 0; kt < 4; ++kt) {
    ah[kt].h[0] = *(const v8h*)&hb[lm * HH + kt * 32 + hi * 8];
    ah[kt].h[1] = *(const v8h*)&hb[lm * HH + kt * 32 + 16 + hi * 8];
  }

  // GEMM2 slice: one N-tile of h@W2 + b2
  v8f acc = {};
#pragma unroll
  for (int kt = 0; kt < 4; ++kt) {
    V16 bf;
    const _Float16* p = &w2frag[((wave * 4 + kt) * 32 + lane) * 16];
    bf.h[0] = *(const v8h*)p;
    bf.h[1] = *(const v8h*)(p + 8);
    acc = __builtin_amdgcn_wmma_f32_16x16x32_f16(
        false, ah[kt].v, false, bf.v, (short)0, acc, false, false);
  }
  const float bv = b2s[wave * 16 + lm];
#pragma unroll
  for (int j = 0; j < 8; ++j) k[j] = acc[j] + bv;
}

__global__ __launch_bounds__(BLOCK) void node_rk4_kernel(
    const float* __restrict__ z0, const float* __restrict__ tgrid,
    const float* __restrict__ W1, const float* __restrict__ b1,
    const float* __restrict__ W2, const float* __restrict__ b2,
    float* __restrict__ out, int Bsz, int T) {
  // Weights pre-packed in exact WMMA B-operand per-lane layout (ISA 7.12.2):
  // lane holds column n = nt*16 + (lane&15); halves 0..7 / 8..15 hold the two
  // K-octets whose base depends on the lane half.
  __shared__ __align__(16) _Float16 w1frag[8 * 2 * 32 * 16];  // 16 KB
  __shared__ __align__(16) _Float16 w2frag[4 * 4 * 32 * 16];  // 16 KB
  __shared__ float b1s[HH];
  __shared__ float b2s[DD];
  __shared__ __align__(16) _Float16 zb[ROWS * DD];  // 2 KB
  __shared__ __align__(16) _Float16 hb[ROWS * HH];  // 4 KB

  const int tid = threadIdx.x;

  for (int idx = tid; idx < 8 * 2 * 32 * 16; idx += BLOCK) {
    const int i = idx & 15;
    const int lane = (idx >> 4) & 31;
    const int kt = (idx >> 9) & 1;
    const int nt = idx >> 10;
    const int n = nt * 16 + (lane & 15);
    const int kbase = (i < 8) ? ((lane < 16) ? 0 : 8) : ((lane < 16) ? 16 : 24);
    const int kk = kt * 32 + kbase + (i & 7);
    w1frag[idx] = (_Float16)W1[kk * HH + n];
  }
  for (int idx = tid; idx < 4 * 4 * 32 * 16; idx += BLOCK) {
    const int i = idx & 15;
    const int lane = (idx >> 4) & 31;
    const int kt = (idx >> 9) & 3;
    const int nt = idx >> 11;
    const int n = nt * 16 + (lane & 15);
    const int kbase = (i < 8) ? ((lane < 16) ? 0 : 8) : ((lane < 16) ? 16 : 24);
    const int kk = kt * 32 + kbase + (i & 7);
    w2frag[idx] = (_Float16)W2[kk * DD + n];
  }
  for (int i = tid; i < HH; i += BLOCK) b1s[i] = b1[i];
  for (int i = tid; i < DD; i += BLOCK) b2s[i] = b2[i];
  __syncthreads();

  const int lane = tid & 31;
  const int wave = tid >> 5;
  const int lm = lane & 15;
  const int hi = lane >> 4;
  const int r0 = blockIdx.x * ROWS;
  const int col0 = wave * 16 + lm;

  // This wave owns one 16-column tile of z in C/D register layout.
  float z[8];
#pragma unroll
  for (int j = 0; j < 8; ++j) {
    const int row = r0 + j + hi * 8;
    z[j] = z0[(size_t)row * DD + col0];
    out[(size_t)row * DD + col0] = z[j];
  }

  float k[8], acc[8];
  const int nsteps = T - 1;
  for (int s = 0; s < nsteps; ++s) {
    const float dt = tgrid[s + 1] - tgrid[s];

    stage(zb, lane, wave, z, z, 0.0f);                       // y = z
    feval(zb, hb, w1frag, w2frag, b1s, b2s, lane, wave, k);  // k1
#pragma unroll
    for (int j = 0; j < 8; ++j) acc[j] = k[j];

    stage(zb, lane, wave, z, k, 0.5f * dt);                  // y = z + dt/2*k1
    feval(zb, hb, w1frag, w2frag, b1s, b2s, lane, wave, k);  // k2
#pragma unroll
    for (int j = 0; j < 8; ++j) acc[j] += 2.0f * k[j];

    stage(zb, lane, wave, z, k, 0.5f * dt);                  // y = z + dt/2*k2
    feval(zb, hb, w1frag, w2frag, b1s, b2s, lane, wave, k);  // k3
#pragma unroll
    for (int j = 0; j < 8; ++j) acc[j] += 2.0f * k[j];

    stage(zb, lane, wave, z, k, dt);                         // y = z + dt*k3
    feval(zb, hb, w1frag, w2frag, b1s, b2s, lane, wave, k);  // k4

    const float c = dt * (1.0f / 6.0f);
    float* outp = out + (size_t)(s + 1) * Bsz * DD;
#pragma unroll
    for (int j = 0; j < 8; ++j) {
      z[j] += c * (acc[j] + k[j]);
      outp[(size_t)(r0 + j + hi * 8) * DD + col0] = z[j];
    }
  }
}

extern "C" void kernel_launch(void* const* d_in, const int* in_sizes, int n_in,
                              void* d_out, int out_size, void* d_ws,
                              size_t ws_size, hipStream_t stream) {
  (void)n_in; (void)out_size; (void)d_ws; (void)ws_size;
  const float* z0 = (const float*)d_in[0];
  const float* t  = (const float*)d_in[1];
  const float* W1 = (const float*)d_in[2];
  const float* b1 = (const float*)d_in[3];
  const float* W2 = (const float*)d_in[4];
  const float* b2 = (const float*)d_in[5];
  float* out = (float*)d_out;
  const int T = in_sizes[1];
  const int Bsz = in_sizes[0] / DD;
  const int grid = Bsz / ROWS;  // 4096/16 = 256 blocks, 4 waves each
  node_rk4_kernel<<<grid, BLOCK, 0, stream>>>(z0, t, W1, b1, W2, b2, out, Bsz, T);
}
